// SelfAttention_82240033784309
// MI455X (gfx1250) — compile-verified, matrix operand order checked by reading, not run on hardware
//
#include <hip/hip_runtime.h>

// ---------------------------------------------------------------------------
// Fused Swin-style window attention for gfx1250 (MI455X).
// One workgroup (256 threads = 8 wave32) per window; all GEMMs via
// v_wmma_f32_16x16x32_f16, intermediates staged in LDS.
// ---------------------------------------------------------------------------

typedef __attribute__((ext_vector_type(16))) _Float16 v16h;
typedef __attribute__((ext_vector_type(8)))  float    v8f;

#define LDX   136   // halfs per row for 64x128 f16 tiles (+8 pad)
#define LDA   68    // halfs per row for 64x64 f16 prob tiles (+4 pad)
#define LDVT  72    // halfs per row for V-transposed (64 cols + 8 pad)
#define LDSS  68    // floats per row for score tiles (+4 pad)

// dynamic-LDS byte offsets
#define OFF_X  0        // sX 64x136 f16 (17408 B)   -- overlaid by sA later
#define OFF_P  17408    // sP 64x136 f16 (17408 B)   -- overlaid by sA later
#define OFF_K  34816    // sK 64x136 f16 (17408 B)
#define OFF_Q  52224    // sQ 64x136 f16 (17408 B)   -- reused as sZ
#define OFF_VT 69632    // sVt 128x72 f16 (18432 B)
#define OFF_S  88064    // sS 4x64x68 f32 (69632 B)
#define SMEM_BYTES 157696

// workspace byte offsets (filled by prep kernel)
#define WS_WKV 0        // 256x128 f16 (65536 B)
#define WS_WQ  65536    // 128x128 f16 (32768 B)
#define WS_WP  98304    // 128x128 f16 (32768 B)
#define WS_RPB 131072   // [4][64][64] f32 (65536 B)

__device__ __forceinline__ v8f vzero() {
  v8f z = {0.f, 0.f, 0.f, 0.f, 0.f, 0.f, 0.f, 0.f};
  return z;
}

// Load a 16x32 K-major operand fragment (A, or B given as B^T rows).
// p points at element [tile_row 0][k 0]; ld = row stride in halfs.
// Lane L<16: row L, K chunks {0..7, 16..23}; lane L>=16: row L-16, {8..15, 24..31}.
__device__ __forceinline__ v16h ldfrag(const _Float16* p, int ld, int lane) {
  int r  = lane & 15;
  int kb = (lane >> 4) << 3;
  const _Float16* q = p + r * ld + kb;
  v16h f;
#pragma unroll
  for (int j = 0; j < 8; ++j) { f[j] = q[j]; f[j + 8] = q[j + 16]; }
  return f;
}

__device__ __forceinline__ v8f wmma32(v16h a, v16h b, v8f c) {
  return __builtin_amdgcn_wmma_f32_16x16x32_f16(
      /*neg_a=*/false, a, /*neg_b=*/false, b,
      /*c_mod=*/(short)0, c, /*reuse_a=*/false, /*reuse_b=*/false);
}

// ---------------------------------------------------------------------------
// Pre-pass: f16 weight copies + relative-position-bias gather into workspace.
// ---------------------------------------------------------------------------
__global__ __launch_bounds__(256) void prep_kernel(
    const float* __restrict__ Wkv, const float* __restrict__ Wq,
    const float* __restrict__ Wproj, const float* __restrict__ bt,
    char* __restrict__ ws) {
  int t = blockIdx.x * 256 + threadIdx.x;
  _Float16* wkv16 = (_Float16*)(ws + WS_WKV);
  _Float16* wq16  = (_Float16*)(ws + WS_WQ);
  _Float16* wp16  = (_Float16*)(ws + WS_WP);
  float*    rpb   = (float*)(ws + WS_RPB);
  if (t < 32768) {
    wkv16[t] = (_Float16)Wkv[t];
  } else if (t < 49152) {
    int i = t - 32768; wq16[i] = (_Float16)Wq[i];
  } else if (t < 65536) {
    int i = t - 49152; wp16[i] = (_Float16)Wproj[i];
  } else if (t < 81920) {
    int d = t - 65536;                 // [h][n][m]
    int h = d >> 12, n = (d >> 6) & 63, m = d & 63;
    int rel = ((n >> 3) - (m >> 3) + 7) * 15 + ((n & 7) - (m & 7) + 7);
    rpb[d] = bt[rel * 4 + h];
  }
}

// ---------------------------------------------------------------------------
// Main fused attention kernel: one block per window.
// ---------------------------------------------------------------------------
__global__ __launch_bounds__(256) void attn_kernel(
    const float* __restrict__ windows, const float* __restrict__ prior,
    const float* __restrict__ mask, const float* __restrict__ b_proj,
    const char* __restrict__ ws, float* __restrict__ out) {
  extern __shared__ char smem[];
  _Float16* sX  = (_Float16*)(smem + OFF_X);
  _Float16* sP  = (_Float16*)(smem + OFF_P);
  _Float16* sA  = (_Float16*)(smem + OFF_X);   // overlays sX+sP after softmax
  _Float16* sK  = (_Float16*)(smem + OFF_K);
  _Float16* sQ  = (_Float16*)(smem + OFF_Q);
  _Float16* sZ  = (_Float16*)(smem + OFF_Q);   // reuses sQ after GEMM3
  _Float16* sVt = (_Float16*)(smem + OFF_VT);
  float*    sS  = (float*)(smem + OFF_S);

  const _Float16* wkv16 = (const _Float16*)(ws + WS_WKV);
  const _Float16* wq16  = (const _Float16*)(ws + WS_WQ);
  const _Float16* wp16  = (const _Float16*)(ws + WS_WP);
  const float*    rpb   = (const float*)(ws + WS_RPB);

  const int b    = blockIdx.x;
  const int t    = threadIdx.x;
  const int wv   = t >> 5;
  const int lane = t & 31;
  const int l15  = lane & 15;
  const int mrb  = (lane >> 4) << 3;   // C/D row base per lane half

  // ---- Stage inputs: f32 -> f16 LDS (fold 1/sqrt(hd) into P) -------------
  {
    const float4* wsrc = (const float4*)(windows + (size_t)b * 8192);
    const float4* psrc = (const float4*)(prior + (size_t)b * 8192);
    const float qs = 0.17677669529663687f;   // 1/sqrt(32)
#pragma unroll
    for (int it = 0; it < 8; ++it) {
      int i = t + it * 256;
      float4 w4 = wsrc[i];
      float4 p4 = psrc[i];
      int e = i << 2;
      int n = e >> 7, c = e & 127;
      _Float16* dx = sX + n * LDX + c;
      dx[0] = (_Float16)w4.x; dx[1] = (_Float16)w4.y;
      dx[2] = (_Float16)w4.z; dx[3] = (_Float16)w4.w;
      _Float16* dp = sP + n * LDX + c;
      dp[0] = (_Float16)(p4.x * qs); dp[1] = (_Float16)(p4.y * qs);
      dp[2] = (_Float16)(p4.z * qs); dp[3] = (_Float16)(p4.w * qs);
    }
  }
  __syncthreads();

  // ---- GEMM1: KV = X @ Wkv^T (64x256, K=128). K -> sK, V -> sVt (transposed)
  {
    int r = wv & 3, hf = wv >> 2;          // hf=0 -> K half, hf=1 -> V half
    v8f acc[8];
#pragma unroll
    for (int j = 0; j < 8; ++j) acc[j] = vzero();
#pragma unroll
    for (int kk = 0; kk < 4; ++kk) {
      v16h a = ldfrag(sX + (r * 16) * LDX + kk * 32, LDX, lane);
#pragma unroll
      for (int j = 0; j < 8; ++j) {
        int n0 = (hf * 8 + j) * 16;
        v16h bf = ldfrag(wkv16 + n0 * 128 + kk * 32, 128, lane);
        acc[j] = wmma32(a, bf, acc[j]);
      }
    }
#pragma unroll
    for (int j = 0; j < 8; ++j) {
      int n0 = (hf * 8 + j) * 16 + l15;
#pragma unroll
      for (int i = 0; i < 8; ++i) {
        int row = r * 16 + mrb + i;
        if (hf == 0) sK[row * LDX + n0] = (_Float16)acc[j][i];
        else         sVt[(n0 - 128) * LDVT + row] = (_Float16)acc[j][i];
      }
    }
  }

  // ---- GEMM2: Q = P_scaled @ Wq^T (64x128, K=128) -> sQ ------------------
  {
    int r = wv & 3, jb = (wv >> 2) * 4;
    v8f acc[4];
#pragma unroll
    for (int j = 0; j < 4; ++j) acc[j] = vzero();
#pragma unroll
    for (int kk = 0; kk < 4; ++kk) {
      v16h a = ldfrag(sP + (r * 16) * LDX + kk * 32, LDX, lane);
#pragma unroll
      for (int jj = 0; jj < 4; ++jj) {
        v16h bf = ldfrag(wq16 + (jb + jj) * 16 * 128 + kk * 32, 128, lane);
        acc[jj] = wmma32(a, bf, acc[jj]);
      }
    }
#pragma unroll
    for (int jj = 0; jj < 4; ++jj) {
      int o = (jb + jj) * 16 + l15;
#pragma unroll
      for (int i = 0; i < 8; ++i)
        sQ[(r * 16 + mrb + i) * LDX + o] = (_Float16)acc[jj][i];
    }
  }
  __syncthreads();

  // ---- GEMM3: S_h = Q_h @ K_h^T + rpb + mask -> sS (f32) -----------------
  {
    int h = wv & 3, rb = (wv >> 2) * 2;
    const float* mk = mask + (size_t)(b & 1023) * 4096;
    const float* rp = rpb + h * 4096;
#pragma unroll
    for (int rr = 0; rr < 2; ++rr) {
      int r = rb + rr;
      v16h a = ldfrag(sQ + (r * 16) * LDX + h * 32, LDX, lane);
#pragma unroll
      for (int ct = 0; ct < 4; ++ct) {
        v16h bf = ldfrag(sK + (ct * 16) * LDX + h * 32, LDX, lane);
        v8f c = wmma32(a, bf, vzero());
        int m = ct * 16 + l15;
#pragma unroll
        for (int i = 0; i < 8; ++i) {
          int n = r * 16 + mrb + i;
          sS[(h * 64 + n) * LDSS + m] = c[i] + rp[n * 64 + m] + mk[n * 64 + m];
        }
      }
    }
  }
  __syncthreads();

  // ---- Softmax: 256 threads <-> 4 heads x 64 rows; probs f16 -> sA -------
  {
    int h = t >> 6, n = t & 63;
    const float* rowp = sS + (h * 64 + n) * LDSS;
    float mx = -3.0e38f;
#pragma unroll
    for (int m = 0; m < 64; ++m) mx = fmaxf(mx, rowp[m]);
    _Float16* arow = sA + (h * 64 + n) * LDA;
    float sum = 0.f;
#pragma unroll
    for (int m = 0; m < 64; ++m) {
      float e = __expf(rowp[m] - mx);
      sum += e;
      arow[m] = (_Float16)e;
    }
    float inv = 1.f / sum;
#pragma unroll
    for (int m = 0; m < 64; ++m)
      arow[m] = (_Float16)((float)arow[m] * inv);
  }
  __syncthreads();

  // ---- GEMM4: Z_h = A_h @ V_h (64x32, K=64) -> sZ ------------------------
  {
    int h = wv & 3, rb = (wv >> 2) * 2;
#pragma unroll
    for (int rr = 0; rr < 2; ++rr) {
      int r = rb + rr;
      v16h a0 = ldfrag(sA + (h * 64 + r * 16) * LDA + 0,  LDA, lane);
      v16h a1 = ldfrag(sA + (h * 64 + r * 16) * LDA + 32, LDA, lane);
#pragma unroll
      for (int ct = 0; ct < 2; ++ct) {
        v16h b0 = ldfrag(sVt + (h * 32 + ct * 16) * LDVT + 0,  LDVT, lane);
        v16h b1 = ldfrag(sVt + (h * 32 + ct * 16) * LDVT + 32, LDVT, lane);
        v8f c = wmma32(a0, b0, vzero());
        c = wmma32(a1, b1, c);
        int d = ct * 16 + l15;
#pragma unroll
        for (int i = 0; i < 8; ++i)
          sZ[(r * 16 + mrb + i) * LDX + h * 32 + d] = (_Float16)c[i];
      }
    }
  }
  __syncthreads();

  // ---- GEMM5: Out = Z @ Wproj^T + b_proj (64x128, K=128) -> global -------
  {
    int r = wv & 3, jb = (wv >> 2) * 4;
    v16h a[4];
#pragma unroll
    for (int kk = 0; kk < 4; ++kk)
      a[kk] = ldfrag(sZ + (r * 16) * LDX + kk * 32, LDX, lane);
    float* op = out + (size_t)b * 8192;
#pragma unroll
    for (int jj = 0; jj < 4; ++jj) {
      int j = jb + jj;
      v8f c = vzero();
#pragma unroll
      for (int kk = 0; kk < 4; ++kk) {
        v16h bf = ldfrag(wp16 + j * 16 * 128 + kk * 32, 128, lane);
        c = wmma32(a[kk], bf, c);
      }
      int o = j * 16 + l15;
      float bias = b_proj[o];
#pragma unroll
      for (int i = 0; i < 8; ++i)
        op[(r * 16 + mrb + i) * 128 + o] = c[i] + bias;
    }
  }
}

// ---------------------------------------------------------------------------
extern "C" void kernel_launch(void* const* d_in, const int* in_sizes, int n_in,
                              void* d_out, int out_size, void* d_ws, size_t ws_size,
                              hipStream_t stream) {
  (void)in_sizes; (void)n_in; (void)out_size; (void)ws_size;
  const float* windows = (const float*)d_in[0];
  const float* prior   = (const float*)d_in[1];
  const float* mask    = (const float*)d_in[2];
  const float* W_kv    = (const float*)d_in[3];
  const float* W_q     = (const float*)d_in[4];
  const float* bt      = (const float*)d_in[5];
  const float* W_proj  = (const float*)d_in[6];
  const float* b_proj  = (const float*)d_in[7];
  float* out = (float*)d_out;
  char*  ws  = (char*)d_ws;

  // allow >64KB dynamic LDS (idempotent, deterministic)
  (void)hipFuncSetAttribute((const void*)attn_kernel,
                            hipFuncAttributeMaxDynamicSharedMemorySize,
                            SMEM_BYTES);

  prep_kernel<<<320, 256, 0, stream>>>(W_kv, W_q, W_proj, bt, ws);
  attn_kernel<<<8192, 256, SMEM_BYTES, stream>>>(windows, prior, mask, b_proj,
                                                 ws, out);
}
